// SpectralGCN_5755256177387
// MI455X (gfx1250) — matrix operand order repaired and checked by reading, not compiled, not run on hardware
//
#include <hip/hip_runtime.h>

// ---------------------------------------------------------------------------
// SpectralGCN: out = U @ diag(g) @ (U^T @ x),  N=8192, F=256, NUM_BINS=15
// MI455X (gfx1250) strategy:
//   * bf16 V_WMMA_F32_16X16X32_BF16 for both 8192x8192x256 GEMMs (f32 acc)
//   * prep pass builds bf16 U, bf16 U^T and bf16 x^T so BOTH GEMMs stage
//     A[m][k] and B^T[n][k] tiles with contiguous 16B chunks
//   * tile staging uses GLOBAL_LOAD_ASYNC_TO_LDS_B128 (ASYNCcnt) — no VGPR
//     round-trip, no LDS scatter; double-buffered, BK=64 (one barrier per
//     8 WMMAs/wave); fragments hoisted so ds_load_b128s pipeline
// ---------------------------------------------------------------------------

typedef __attribute__((ext_vector_type(16))) __bf16 v16bf;
typedef __attribute__((ext_vector_type(8)))  float  v8f;

#define NN 8192
#define FF 256

__device__ __forceinline__ unsigned short f2bf(float f) {
    // round-to-nearest-even fp32 -> bf16
    unsigned int u = __float_as_uint(f);
    unsigned int r = (u + 0x7FFFu + ((u >> 16) & 1u)) >> 16;
    return (unsigned short)r;
}

// -------------------------- prep kernels -----------------------------------

__global__ __launch_bounds__(256) void filter_lut_kernel(
    const float* __restrict__ lam, const float* __restrict__ fp,
    float* __restrict__ g, int n) {
    int i = blockIdx.x * 256 + threadIdx.x;
    if (i < n) {
        // bin_width = 2/(15-1) = 1/7  ->  idx = trunc(lam * 7), clamp [0,14]
        int idx = (int)(lam[i] * 7.0f);
        idx = idx < 0 ? 0 : (idx > 14 ? 14 : idx);
        g[i] = fp[idx];
    }
}

__global__ __launch_bounds__(256) void cvt_f32_bf16_kernel(
    const float* __restrict__ src, unsigned short* __restrict__ dst, int n8) {
    int i = blockIdx.x * 256 + threadIdx.x;
    if (i >= n8) return;
    float4 a = reinterpret_cast<const float4*>(src)[2 * (long long)i];
    float4 b = reinterpret_cast<const float4*>(src)[2 * (long long)i + 1];
    union { uint4 v; unsigned short u[8]; } o;
    o.u[0] = f2bf(a.x); o.u[1] = f2bf(a.y); o.u[2] = f2bf(a.z); o.u[3] = f2bf(a.w);
    o.u[4] = f2bf(b.x); o.u[5] = f2bf(b.y); o.u[6] = f2bf(b.z); o.u[7] = f2bf(b.w);
    reinterpret_cast<uint4*>(dst)[i] = o.v;
}

// dst[c][r] = bf16(src[r][c]); src is R x C f32, dst is C x R bf16.
// 32x32 LDS tile, block (32,8): coalesced reads and writes.
__global__ __launch_bounds__(256) void transpose_cvt_kernel(
    const float* __restrict__ src, unsigned short* __restrict__ dst,
    int R, int C) {
    __shared__ float tile[32][33];
    const int c0 = blockIdx.x * 32, r0 = blockIdx.y * 32;
    const int tx = threadIdx.x, ty = threadIdx.y;
#pragma unroll
    for (int j = 0; j < 4; ++j)
        tile[ty + 8 * j][tx] = src[(size_t)(r0 + ty + 8 * j) * C + (c0 + tx)];
    __syncthreads();
#pragma unroll
    for (int j = 0; j < 4; ++j)
        dst[(size_t)(c0 + ty + 8 * j) * R + (r0 + tx)] = f2bf(tile[tx][ty + 8 * j]);
}

// ------------------------------ bf16 WMMA GEMM -----------------------------
// C[M=8192, N=FF] = A[8192 x 8192] * B[8192 x FF]
//   A   given row-major [m][k] (ld NN)
//   B   given pre-transposed [n][k] (ld NN)  -> both tiles stage contiguously
//   SCALE: epilogue scales row m by g[m], writes bf16 C^T[n][m] (contiguous
//          per-lane 16B stores); else writes f32 C[m][n].
// Block: 128 threads = 4 wave32. Tile BM=64, BN=64, BK=64, double-buffered.
// Each wave owns a 64x16 strip of C = 4 wmma accumulators; 8 WMMAs / barrier.

#define LDK 72   // padded LDS row stride in elements (144 B, 16B-aligned)

template <bool SCALE>
__global__ __launch_bounds__(128) void gemm_bf16_kernel(
    const unsigned short* __restrict__ Au,   // bf16 [8192][8192]
    const unsigned short* __restrict__ Btu,  // bf16 [FF][8192] (B transposed)
    const float* __restrict__ g,             // row scales (SCALE epilogue)
    unsigned short* __restrict__ outT,       // bf16 [FF][8192] (SCALE)
    float* __restrict__ outf)                // f32  [8192][FF] (!SCALE)
{
    __shared__ __align__(16) __bf16 Asm[2][64][LDK];   // [buf][m][k]
    __shared__ __align__(16) __bf16 Bsm[2][64][LDK];   // [buf][n][k]

    const int tid  = threadIdx.x;
    const int wave = tid >> 5;        // 0..3 -> 16-column strip of C
    const int lane = tid & 31;
    const int kh   = lane >> 4;       // wave half (CDNA5 wmma layouts)
    const int l15  = lane & 15;

    const int n0 = blockIdx.x * 64;   // FF/64 = 4
    const int m0 = blockIdx.y * 64;   // NN/64 = 128

    // LDS byte offsets of the tile bases (low 32 bits of the flat address
    // are the wave-relative LDS offset per the gfx1250 aperture layout).
    const unsigned ldsA = (unsigned)(size_t)(&Asm[0][0][0]);
    const unsigned ldsB = (unsigned)(size_t)(&Bsm[0][0][0]);

    v8f acc[4];
#pragma unroll
    for (int t = 0; t < 4; ++t)
#pragma unroll
        for (int j = 0; j < 8; ++j) acc[t][j] = 0.0f;

    // ---- async staging: 512 chunks of 16B per tile, 4 A + 4 B per thread ---
    const int kc = (tid & 7) * 8;     // 0..56 within BK=64
    auto stage = [&](int k0, int buf) {
        const unsigned bufOff = (unsigned)buf * (unsigned)(64 * LDK * 2);
#pragma unroll
        for (int c = 0; c < 4; ++c) {
            const int row = (tid >> 3) + 16 * c;          // 0..63
            const unsigned lo = bufOff + (unsigned)(row * (LDK * 2) + kc * 2);
            unsigned long long ga = (unsigned long long)(size_t)Au +
                2ull * ((size_t)(m0 + row) * NN + (size_t)(k0 + kc));
            asm volatile("global_load_async_to_lds_b128 %0, %1, off"
                         :: "v"(ldsA + lo), "v"(ga) : "memory");
            unsigned long long gb = (unsigned long long)(size_t)Btu +
                2ull * ((size_t)(n0 + row) * NN + (size_t)(k0 + kc));
            asm volatile("global_load_async_to_lds_b128 %0, %1, off"
                         :: "v"(ldsB + lo), "v"(gb) : "memory");
        }
    };

    // ---- prologue: async-fill buffer 0 ----
    stage(0, 0);
    asm volatile("s_wait_asynccnt 0x0" ::: "memory");
    __syncthreads();

    int buf = 0;
    for (int k0 = 0; k0 < NN; k0 += 64, buf ^= 1) {
        const bool more = (k0 + 64 < NN);
        // fire next tile's async copies into the other buffer (safe: its last
        // readers completed before the previous barrier)
        if (more) stage(k0 + 64, buf ^ 1);

#pragma unroll
        for (int kk = 0; kk < 2; ++kk) {       // two K=32 WMMA steps
            const int kb0 = kk * 32;

            // hoist all fragment loads so the ds_load_b128s pipeline
            // B fragment (32x16): lane -> column n, K = kb0 + 16*half + e
            v16bf bfrag;
#pragma unroll
            for (int e = 0; e < 16; ++e)
                bfrag[e] = Bsm[buf][wave * 16 + l15][kb0 + 16 * kh + e];

            // A fragments (16x32): lane -> row m, K = kb0 + 16*half + e
            v16bf af[4];
#pragma unroll
            for (int mt = 0; mt < 4; ++mt)
#pragma unroll
                for (int p = 0; p < 8; ++p) {
                    int kb = kb0 + 2 * p + 16 * kh;
                    af[mt][2 * p]     = Asm[buf][mt * 16 + l15][kb];
                    af[mt][2 * p + 1] = Asm[buf][mt * 16 + l15][kb + 1];
                }

#pragma unroll
            for (int mt = 0; mt < 4; ++mt)
                acc[mt] = __builtin_amdgcn_wmma_f32_16x16x32_bf16(
                    false, af[mt], false, bfrag, (short)0, acc[mt], false, false);
        }

        if (more) {
            asm volatile("s_wait_asynccnt 0x0" ::: "memory");
            __syncthreads();
        }
    }

    // ---- epilogue: C layout, vgpr j -> M = j + 8*half, lane -> N = l15 ----
    const int n = n0 + wave * 16 + l15;
    if (SCALE) {
        // write C^T[n][m] in bf16 with the g[m] row scale; per-lane the 8
        // accumulator rows are contiguous in m -> one 16B store per tile
#pragma unroll
        for (int mt = 0; mt < 4; ++mt) {
            const int mb = m0 + mt * 16 + 8 * kh;
            union { uint4 v; unsigned short u[8]; } o;
#pragma unroll
            for (int j = 0; j < 8; ++j)
                o.u[j] = f2bf(g[mb + j] * acc[mt][j]);
            *reinterpret_cast<uint4*>(&outT[(size_t)n * NN + mb]) = o.v;
        }
    } else {
#pragma unroll
        for (int mt = 0; mt < 4; ++mt) {
#pragma unroll
            for (int j = 0; j < 8; ++j) {
                int m = m0 + mt * 16 + 8 * kh + j;
                outf[(size_t)m * FF + n] = acc[mt][j];
            }
        }
    }
}

// ------------------------------- launcher ----------------------------------

extern "C" void kernel_launch(void* const* d_in, const int* in_sizes, int n_in,
                              void* d_out, int out_size, void* d_ws, size_t ws_size,
                              hipStream_t stream) {
    (void)in_sizes; (void)n_in; (void)out_size; (void)ws_size;
    const float* x   = (const float*)d_in[0];   // [N,F]
    const float* U   = (const float*)d_in[1];   // [N,N]
    const float* lam = (const float*)d_in[2];   // [N]
    const float* fp  = (const float*)d_in[3];   // [15]
    // d_in[4] edge_index: unused by the reference computation
    float* out = (float*)d_out;                 // [N,F] f32

    // workspace layout (~264 MB)
    char* ws = (char*)d_ws;
    size_t off = 0;
    unsigned short* Ubf  = (unsigned short*)(ws + off); off += (size_t)NN * NN * 2; // 128 MB
    unsigned short* UTbf = (unsigned short*)(ws + off); off += (size_t)NN * NN * 2; // 128 MB
    unsigned short* xTbf = (unsigned short*)(ws + off); off += (size_t)NN * FF * 2; //   4 MB
    unsigned short* TTbf = (unsigned short*)(ws + off); off += (size_t)NN * FF * 2; //   4 MB
    float*          g    = (float*)(ws + off);                                      //  32 KB

    // 1) g[i] = filter_params[clamp(trunc(lam*7), 0, 14)]
    filter_lut_kernel<<<(NN + 255) / 256, 256, 0, stream>>>(lam, fp, g, NN);

    // 2) Ubf = bf16(U)
    {
        int n8U = (NN * (size_t)NN) / 8;  // 8,388,608
        cvt_f32_bf16_kernel<<<(n8U + 255) / 256, 256, 0, stream>>>(U, Ubf, n8U);
    }
    // 3) UTbf = bf16(U^T), xTbf = bf16(x^T)
    transpose_cvt_kernel<<<dim3(NN / 32, NN / 32), dim3(32, 8), 0, stream>>>(
        U, UTbf, NN, NN);
    transpose_cvt_kernel<<<dim3(FF / 32, NN / 32), dim3(32, 8), 0, stream>>>(
        x, xTbf, NN, FF);

    dim3 grid(FF / 64, NN / 64);  // (4, 128)
    dim3 block(128);

    // 4) T^T[n][m] = g[m] * (U^T @ x)[m][n], bf16   (A = U^T, B^T = x^T)
    gemm_bf16_kernel<true><<<grid, block, 0, stream>>>(
        UTbf, xTbf, g, TTbf, nullptr);

    // 5) out = U @ T, f32                           (A = U,   B^T = T^T)
    gemm_bf16_kernel<false><<<grid, block, 0, stream>>>(
        Ubf, TTbf, g, nullptr, out);
}
